// MultiHeadAttention_33698313405256
// MI455X (gfx1250) — matrix operand order repaired and checked by reading, not compiled, not run on hardware
//
#include <hip/hip_runtime.h>
#include <stdint.h>

typedef __attribute__((ext_vector_type(16))) __bf16 v16bf;
typedef __attribute__((ext_vector_type(8)))  float  v8f;

#define BB 4
#define TT 2048
#define DD 1024
#define NH 16
#define DH 64
#define BT (BB*TT)

__device__ __forceinline__ unsigned short f2bf(float f) {
  union { float f; uint32_t u; } c; c.f = f;
  uint32_t u = c.u;
  uint32_t r = (u + 0x7FFFu + ((u >> 16) & 1u)) >> 16;  // RNE
  return (unsigned short)r;
}

union FragBF { uint4 q[2]; v16bf v; };

// ---------------------------------------------------------------------------
// CDNA5 async memory->LDS copy (ASYNCcnt-tracked), 16B per lane-op.
// LDS byte address = low 32 bits of the generic pointer (aperture truncation).
// ---------------------------------------------------------------------------
__device__ __forceinline__ void async_copy_b128(const void* gptr, void* lptr) {
  uint32_t lds = (uint32_t)(uintptr_t)lptr;
  asm volatile("global_load_async_to_lds_b128 %0, %1, off"
               :: "v"(lds), "v"(gptr) : "memory");
}
__device__ __forceinline__ void wait_async_all() {
  asm volatile("s_wait_asynccnt 0x0" ::: "memory");
}

// ---------------------------------------------------------------------------
// fp32 -> bf16 conversion (x)
// ---------------------------------------------------------------------------
__global__ void f32_to_bf16_kernel(const float* __restrict__ in,
                                   unsigned short* __restrict__ out, int n) {
  int i = blockIdx.x * blockDim.x + threadIdx.x;
  int stride = gridDim.x * blockDim.x;
  for (; i < n; i += stride) out[i] = f2bf(in[i]);
}

// fp32 weight (NxK row-major) -> bf16 transposed W^T (KxN row-major)
__global__ void convert_w_transpose_kernel(const float* __restrict__ w,
                                           unsigned short* __restrict__ wt) {
  int i = blockIdx.x * blockDim.x + threadIdx.x;
  int stride = gridDim.x * blockDim.x;
  const int total = DD * DD;
  for (; i < total; i += stride) {
    int k = i >> 10, n = i & 1023;
    wt[i] = f2bf(w[(size_t)n * DD + k]);  // wt[k][n] = w[n][k]
  }
}

// K (BT x D bf16, head-sliced cols) -> KT[(b*NH+h)*64 + d][t]  (per-head d-major)
__global__ void transpose_k_kernel(const unsigned short* __restrict__ Kb,
                                   unsigned short* __restrict__ KT) {
  int i = blockIdx.x * blockDim.x + threadIdx.x;
  int stride = gridDim.x * blockDim.x;
  const int total = BT * DD;
  for (; i < total; i += stride) {
    int row = i >> 10;            // b*TT + t
    int col = i & 1023;           // h*64 + d
    int b = row >> 11, t = row & 2047;
    int h = col >> 6,  d = col & 63;
    KT[((size_t)(b * NH + h) * DH + d) * TT + t] = Kb[i];
  }
}

// ---------------------------------------------------------------------------
// C(MxN) = A(MxK, bf16 row-major) * Wt(KxN, bf16 row-major)
// 128x128 block tile, 8 waves (2x4), each wave 64x32 = 4x2 WMMA tiles.
// Double-buffered LDS; tile k+1 DMA overlaps tile-k WMMAs.
// ---------------------------------------------------------------------------
__device__ __forceinline__ void gemm_issue(const unsigned short* A,
                                           const unsigned short* Wt,
                                           unsigned short* As, unsigned short* Bs,
                                           int m0, int n0, int k0, int N, int K,
                                           int tid) {
#pragma unroll
  for (int it = 0; it < 2; ++it) {
    int i = tid + it * 256;
    int arow = i >> 2, akc = (i & 3) * 8;                 // A: 128 x 32
    async_copy_b128(A + (size_t)(m0 + arow) * K + k0 + akc,
                    As + arow * 32 + akc);
    int bk = i >> 4, bnc = (i & 15) * 8;                  // B: 32 x 128
    async_copy_b128(Wt + (size_t)(k0 + bk) * N + n0 + bnc,
                    Bs + bk * 128 + bnc);
  }
}

template <int OUTF32>
__global__ __launch_bounds__(256)
void gemm_wmma(const unsigned short* __restrict__ A,
               const unsigned short* __restrict__ Wt,
               void* __restrict__ Cout, int M, int N, int K) {
  __shared__ unsigned short As[2][128 * 32];   // [buf][row][k]
  __shared__ unsigned short Bs[2][32 * 128];   // [buf][k][n]
  const int tid  = threadIdx.x;
  const int lane = tid & 31, wave = tid >> 5;
  const int wm = wave >> 2, wn = wave & 3;
  const int m0 = blockIdx.y * 128, n0 = blockIdx.x * 128;
  const int c0 = (lane < 16) ? 0 : 8;   // A-frag K-chunk base / D-row offset
  const int l15 = lane & 15;

  v8f zero = {0.f, 0.f, 0.f, 0.f, 0.f, 0.f, 0.f, 0.f};
  v8f acc[4][2];
  for (int i = 0; i < 4; ++i)
    for (int j = 0; j < 2; ++j) acc[i][j] = zero;

  const int nsteps = K / 32;
  gemm_issue(A, Wt, As[0], Bs[0], m0, n0, 0, N, K, tid);   // prologue

  for (int ks = 0; ks < nsteps; ++ks) {
    const int buf = ks & 1;
    wait_async_all();      // own tile-ks copies complete (only ones in flight)
    __syncthreads();       // publish tile ks; retire readers of buf^1
    if (ks + 1 < nsteps)
      gemm_issue(A, Wt, As[buf ^ 1], Bs[buf ^ 1], m0, n0, (ks + 1) * 32, N, K, tid);

    const unsigned short* as = As[buf];
    const unsigned short* bs = Bs[buf];
#pragma unroll
    for (int mt = 0; mt < 4; ++mt) {
      FragBF a;
      int arow = wm * 64 + mt * 16 + l15;
      a.q[0] = *(const uint4*)(as + arow * 32 + c0);
      a.q[1] = *(const uint4*)(as + arow * 32 + c0 + 16);
#pragma unroll
      for (int nt = 0; nt < 2; ++nt) {
        FragBF b;
        int nb = wn * 32 + nt * 16;
        for (int s = 0; s < 16; ++s)
          ((unsigned short*)&b)[s] = bs[(c0 + (s & 7) + (s >> 3) * 16) * 128 + nb + l15];
        acc[mt][nt] = __builtin_amdgcn_wmma_f32_16x16x32_bf16(
            false, a.v, false, b.v, (short)0, acc[mt][nt], false, false);
      }
    }
  }

#pragma unroll
  for (int mt = 0; mt < 4; ++mt)
#pragma unroll
    for (int nt = 0; nt < 2; ++nt)
#pragma unroll
      for (int r = 0; r < 8; ++r) {
        int row = m0 + wm * 64 + mt * 16 + r + c0;
        int col = n0 + wn * 32 + nt * 16 + l15;
        float v = acc[mt][nt][r];
        if (OUTF32) ((float*)Cout)[(size_t)row * N + col] = v;
        else ((unsigned short*)Cout)[(size_t)row * N + col] = f2bf(v);
      }
}

// ---------------------------------------------------------------------------
// Flash attention: block = 128 query rows of one (b,h); 8 waves x 16 rows.
// Double-buffered K^T / V tiles via async DMA; online softmax in registers;
// P goes D-layout -> A-layout through per-wave LDS scratch.
// ---------------------------------------------------------------------------
__device__ __forceinline__ void flash_issue(const unsigned short* KT,
                                            const unsigned short* Vm,
                                            unsigned short* sKt, unsigned short* sVs,
                                            size_t ktbase, size_t rowbase,
                                            int h, int kv, int tid) {
#pragma unroll
  for (int it = 0; it < 2; ++it) {
    int i = tid + it * 256;
    int rr = i >> 3, cc = (i & 7) * 8;
    async_copy_b128(KT + ktbase + (size_t)rr * TT + kv + cc, sKt + rr * 64 + cc);
    async_copy_b128(Vm + (rowbase + kv + rr) * DD + h * DH + cc, sVs + rr * 64 + cc);
  }
}

__global__ __launch_bounds__(256)
void flash_attn_wmma(const unsigned short* __restrict__ Q,
                     const unsigned short* __restrict__ KT,
                     const unsigned short* __restrict__ Vm,
                     unsigned short* __restrict__ Y) {
  __shared__ unsigned short sKt[2][64 * 64];   // [buf][d][key]
  __shared__ unsigned short sVs[2][64 * 64];   // [buf][key][d]
  __shared__ unsigned short sPs[8 * 16 * 64];  // per-wave P scratch
  const int tid = threadIdx.x, lane = tid & 31, wave = tid >> 5;
  const int q0 = blockIdx.x * 128;
  const int bh = blockIdx.y;
  const int b = bh >> 4, h = bh & 15;
  const size_t rowbase = (size_t)b * TT;
  const size_t ktbase = (size_t)bh * DH * TT;  // KT[(bh*64 + d)*TT + t]
  const int half8 = (lane < 16) ? 0 : 8;
  const int l15 = lane & 15;

  // Q fragments for this wave's 16 rows (d = 64 -> two K-steps)
  FragBF a0, a1;
  {
    size_t qrow = rowbase + q0 + wave * 16 + l15;
    const unsigned short* qp = Q + qrow * DD + h * DH;
    a0.q[0] = *(const uint4*)(qp + half8);
    a0.q[1] = *(const uint4*)(qp + half8 + 16);
    a1.q[0] = *(const uint4*)(qp + 32 + half8);
    a1.q[1] = *(const uint4*)(qp + 32 + half8 + 16);
  }

  v8f zero = {0.f, 0.f, 0.f, 0.f, 0.f, 0.f, 0.f, 0.f};
  v8f o[4];
  for (int i = 0; i < 4; ++i) o[i] = zero;
  float row_m[8], row_l[8];
#pragma unroll
  for (int r = 0; r < 8; ++r) { row_m[r] = -1e30f; row_l[r] = 0.f; }

  const int kv_end = q0 + 128;  // causal
  flash_issue(KT, Vm, sKt[0], sVs[0], ktbase, rowbase, h, 0, tid);  // prologue

  for (int kv = 0; kv < kv_end; kv += 64) {
    const int buf = (kv >> 6) & 1;
    wait_async_all();
    __syncthreads();
    if (kv + 64 < kv_end)
      flash_issue(KT, Vm, sKt[buf ^ 1], sVs[buf ^ 1], ktbase, rowbase, h, kv + 64, tid);
    const unsigned short* kt = sKt[buf];
    const unsigned short* vs = sVs[buf];

    // S = Q K^T over the 64-key tile (4 N-tiles x 2 K-steps)
    v8f s[4];
#pragma unroll
    for (int nt = 0; nt < 4; ++nt) {
      v8f acc_s = zero;
      FragBF bk;
      for (int s = 0; s < 16; ++s)
        ((unsigned short*)&bk)[s] = kt[(half8 + (s & 7) + (s >> 3) * 16) * 64 + nt * 16 + l15];
      acc_s = __builtin_amdgcn_wmma_f32_16x16x32_bf16(
          false, a0.v, false, bk.v, (short)0, acc_s, false, false);
      for (int s = 0; s < 16; ++s)
        ((unsigned short*)&bk)[s] = kt[(32 + half8 + (s & 7) + (s >> 3) * 16) * 64 + nt * 16 + l15];
      acc_s = __builtin_amdgcn_wmma_f32_16x16x32_bf16(
          false, a1.v, false, bk.v, (short)0, acc_s, false, false);
      s[nt] = acc_s;
    }

    // scale + causal mask (D layout: reg r -> row, lane%16 -> col)
    const int qrow_lo = q0 + wave * 16 + half8;
#pragma unroll
    for (int nt = 0; nt < 4; ++nt) {
      int colg = kv + nt * 16 + l15;
#pragma unroll
      for (int r = 0; r < 8; ++r) {
        float sv = s[nt][r] * 0.125f;  // 1/sqrt(64)
        s[nt][r] = (colg <= qrow_lo + r) ? sv : -1e30f;
      }
    }

    // online softmax: row reductions across 16 lanes (halves independent)
    float mnew[8], alpha[8];
#pragma unroll
    for (int r = 0; r < 8; ++r) {
      float mx = fmaxf(fmaxf(s[0][r], s[1][r]), fmaxf(s[2][r], s[3][r]));
#pragma unroll
      for (int sh = 1; sh < 16; sh <<= 1) mx = fmaxf(mx, __shfl_xor(mx, sh, 32));
      mnew[r] = fmaxf(row_m[r], mx);
      alpha[r] = __expf(row_m[r] - mnew[r]);
      row_m[r] = mnew[r];
    }
    float rsum[8];
#pragma unroll
    for (int r = 0; r < 8; ++r) rsum[r] = 0.f;
#pragma unroll
    for (int nt = 0; nt < 4; ++nt)
#pragma unroll
      for (int r = 0; r < 8; ++r) {
        float p = __expf(s[nt][r] - mnew[r]);
        s[nt][r] = p;
        rsum[r] += p;
      }
#pragma unroll
    for (int r = 0; r < 8; ++r) {
      float t = rsum[r];
#pragma unroll
      for (int sh = 1; sh < 16; sh <<= 1) t += __shfl_xor(t, sh, 32);
      row_l[r] = row_l[r] * alpha[r] + t;
#pragma unroll
      for (int ot = 0; ot < 4; ++ot) o[ot][r] *= alpha[r];
    }

    // P: D-layout -> row-major bf16 in per-wave LDS (A-layout reload below)
    unsigned short* pw = sPs + wave * (16 * 64);
#pragma unroll
    for (int nt = 0; nt < 4; ++nt)
#pragma unroll
      for (int r = 0; r < 8; ++r)
        pw[(r + half8) * 64 + nt * 16 + l15] = f2bf(s[nt][r]);

    // O += P * V  (K-dim = 64 keys -> two K-steps; 4 d-tiles)
#pragma unroll
    for (int ks = 0; ks < 2; ++ks) {
      FragBF pa;
      const unsigned short* pr = pw + l15 * 64 + ks * 32 + half8;
      pa.q[0] = *(const uint4*)(pr);
      pa.q[1] = *(const uint4*)(pr + 16);
#pragma unroll
      for (int ot = 0; ot < 4; ++ot) {
        FragBF bv;
        for (int s = 0; s < 16; ++s)
          ((unsigned short*)&bv)[s] = vs[(ks * 32 + half8 + (s & 7) + (s >> 3) * 16) * 64 + ot * 16 + l15];
        o[ot] = __builtin_amdgcn_wmma_f32_16x16x32_bf16(
            false, pa.v, false, bv.v, (short)0, o[ot], false, false);
      }
    }
  }

  // normalize and store Y
#pragma unroll
  for (int r = 0; r < 8; ++r) {
    float inv = 1.f / row_l[r];
    size_t yrow = rowbase + q0 + wave * 16 + r + half8;
#pragma unroll
    for (int ot = 0; ot < 4; ++ot)
      Y[yrow * DD + h * DH + ot * 16 + l15] = f2bf(o[ot][r] * inv);
  }
}

// ---------------------------------------------------------------------------
extern "C" void kernel_launch(void* const* d_in, const int* in_sizes, int n_in,
                              void* d_out, int out_size, void* d_ws, size_t ws_size,
                              hipStream_t stream) {
  (void)in_sizes; (void)n_in; (void)out_size; (void)ws_size;
  const float* x  = (const float*)d_in[0];
  const float* wq = (const float*)d_in[1];
  const float* wk = (const float*)d_in[2];
  const float* wv = (const float*)d_in[3];
  const float* wo = (const float*)d_in[4];

  char* ws = (char*)d_ws;
  const size_t MB = 1024 * 1024;
  unsigned short* xb  = (unsigned short*)(ws + 0 * MB);   // 16 MB
  unsigned short* wqt = (unsigned short*)(ws + 16 * MB);  //  2 MB each (W^T)
  unsigned short* wkt = (unsigned short*)(ws + 18 * MB);
  unsigned short* wvt = (unsigned short*)(ws + 20 * MB);
  unsigned short* wot = (unsigned short*)(ws + 22 * MB);
  unsigned short* Qb  = (unsigned short*)(ws + 24 * MB);  // 16 MB
  unsigned short* Kb  = (unsigned short*)(ws + 40 * MB);  // 16 MB
  unsigned short* Vb  = (unsigned short*)(ws + 56 * MB);  // 16 MB
  unsigned short* Yb  = (unsigned short*)(ws + 72 * MB);  // 16 MB
  unsigned short* KTb = (unsigned short*)(ws + 88 * MB);  // 16 MB (per-head K^T)

  const int nx = BT * DD;
  f32_to_bf16_kernel<<<1024, 256, 0, stream>>>(x, xb, nx);
  convert_w_transpose_kernel<<<512, 256, 0, stream>>>(wq, wqt);
  convert_w_transpose_kernel<<<512, 256, 0, stream>>>(wk, wkt);
  convert_w_transpose_kernel<<<512, 256, 0, stream>>>(wv, wvt);
  convert_w_transpose_kernel<<<512, 256, 0, stream>>>(wo, wot);

  dim3 g1(DD / 128, BT / 128);  // (8, 64)
  gemm_wmma<0><<<g1, 256, 0, stream>>>(xb, wqt, Qb, BT, DD, DD);
  gemm_wmma<0><<<g1, 256, 0, stream>>>(xb, wkt, Kb, BT, DD, DD);
  gemm_wmma<0><<<g1, 256, 0, stream>>>(xb, wvt, Vb, BT, DD, DD);

  transpose_k_kernel<<<1024, 256, 0, stream>>>(Kb, KTb);

  dim3 g2(TT / 128, BB * NH);   // (16, 64)
  flash_attn_wmma<<<g2, 256, 0, stream>>>(Qb, KTb, Vb, Yb);

  gemm_wmma<1><<<g1, 256, 0, stream>>>(Yb, wot, d_out, BT, DD, DD);
}
